// MultiHeadAttention_8426725835073
// MI455X (gfx1250) — compile-verified, matrix operand order checked by reading, not run on hardware
//
#include <hip/hip_runtime.h>
#include <hip/hip_bf16.h>

// ---------------------------------------------------------------------------
// MHA for MI455X (gfx1250): bf16 WMMA GEMMs + flash-attention with
// double-buffered async-to-LDS K/V staging (ASYNCcnt path).
// ---------------------------------------------------------------------------

typedef __bf16 bf16_t;
typedef __attribute__((ext_vector_type(16))) __bf16 bf16x16;
typedef __attribute__((ext_vector_type(2)))  __bf16 bf16x2;
typedef __attribute__((ext_vector_type(8)))  float  f32x8;
typedef __attribute__((ext_vector_type(2)))  float  f32x2;

#define S_LEN   2048
#define D_MODEL 1024
#define NHEADS  16
#define HEAD_D  64
#define NEG_INF (-1e20f)

__device__ __forceinline__ bf16_t f2bf(float f) {
  unsigned int u = __float_as_uint(f);
  u += 0x7FFFu + ((u >> 16) & 1u);          // round-to-nearest-even
  unsigned short h = (unsigned short)(u >> 16);
  union { unsigned short s; bf16_t b; } c; c.s = h; return c.b;
}

__device__ __forceinline__ f32x8 wmma_bf16(bf16x16 a, bf16x16 b, f32x8 c) {
  return __builtin_amdgcn_wmma_f32_16x16x32_bf16(
      false, a, false, b, (short)0, c, false, false);
}

// xor-butterfly max within each 16-lane half using v_permlane16_b32.
__device__ __forceinline__ float permlane_xor_f(float v, unsigned lo, unsigned hi) {
  unsigned u = __float_as_uint(v);
  unsigned r = __builtin_amdgcn_permlane16(u, u, lo, hi, false, false);
  return __uint_as_float(r);
}
__device__ __forceinline__ float halfwave_max16(float v) {
  v = fmaxf(v, permlane_xor_f(v, 0x67452301u, 0xEFCDAB89u)); // xor 1
  v = fmaxf(v, permlane_xor_f(v, 0x54761032u, 0xDCFE98BAu)); // xor 2
  v = fmaxf(v, permlane_xor_f(v, 0x32107654u, 0xBA98FEDCu)); // xor 4
  v = fmaxf(v, permlane_xor_f(v, 0xFEDCBA98u, 0x76543210u)); // xor 8
  return v;
}

// Async DMA: global -> LDS, 16B per lane, tracked by ASYNCcnt.
__device__ __forceinline__ void async_b128(unsigned lds_off, const bf16_t* g) {
  asm volatile("global_load_async_to_lds_b128 %0, %1, off"
               :: "v"(lds_off), "v"((unsigned long long)(size_t)g)
               : "memory");
}

// Load a 16x32 bf16 WMMA fragment from row-major [rows][ld] storage
// (global or LDS).  A-fragment and [N][K]-stored B-fragment layouts match.
__device__ __forceinline__ bf16x16 load_frag(const bf16_t* base,
                                             int ld, int row0, int col0, int lane) {
  const int r = row0 + (lane & 15);
  const int c = col0 + ((lane >> 4) << 3);
  const bf16_t* p = base + (size_t)r * ld + c;
  union { bf16x16 f; uint4 u[2]; } x;
  x.u[0] = *reinterpret_cast<const uint4*>(p);
  x.u[1] = *reinterpret_cast<const uint4*>(p + 16);
  return x.f;
}

// ---------------------------------------------------------------------------
// f32 -> bf16 conversion (vectorized x4)
// ---------------------------------------------------------------------------
__global__ __launch_bounds__(256) void cvt_kernel(const float* __restrict__ src,
                                                  bf16_t* __restrict__ dst, int n4) {
  int i = blockIdx.x * 256 + threadIdx.x;
  if (i >= n4) return;
  const float4 v = reinterpret_cast<const float4*>(src)[i];
  union { bf16_t b[4]; uint2 u; } o;
  o.b[0] = f2bf(v.x); o.b[1] = f2bf(v.y); o.b[2] = f2bf(v.z); o.b[3] = f2bf(v.w);
  reinterpret_cast<uint2*>(dst)[i] = o.u;
}

// ---------------------------------------------------------------------------
// GEMM: out = A[M=8192][1024] * W^T + bias  (W stored [out_feat][in_feat])
// Block = 8 waves; block tile 128x128; wave tile 32x64 (2x4 WMMA)
// ---------------------------------------------------------------------------
template <int MODE>
__global__ __launch_bounds__(256) void gemm_bf16_kernel(
    const bf16_t* __restrict__ A, const bf16_t* __restrict__ W,
    const float* __restrict__ bias, bf16_t* __restrict__ outb,
    float* __restrict__ outf) {
  const int lane = threadIdx.x & 31;
  const int wave = threadIdx.x >> 5;
  const int m0 = blockIdx.x * 128 + (wave >> 1) * 32;
  const int n0 = blockIdx.y * 128 + (wave & 1) * 64;

  f32x8 acc[2][4];
#pragma unroll
  for (int mt = 0; mt < 2; ++mt)
#pragma unroll
    for (int nt = 0; nt < 4; ++nt)
      acc[mt][nt] = (f32x8){0.f,0.f,0.f,0.f,0.f,0.f,0.f,0.f};

  for (int kk = 0; kk < D_MODEL; kk += 32) {
    const bf16x16 a0 = load_frag(A, D_MODEL, m0,      kk, lane);
    const bf16x16 a1 = load_frag(A, D_MODEL, m0 + 16, kk, lane);
    bf16x16 bfrag[4];
#pragma unroll
    for (int nt = 0; nt < 4; ++nt)
      bfrag[nt] = load_frag(W, D_MODEL, n0 + nt * 16, kk, lane);
#pragma unroll
    for (int nt = 0; nt < 4; ++nt) {
      acc[0][nt] = wmma_bf16(a0, bfrag[nt], acc[0][nt]);
      acc[1][nt] = wmma_bf16(a1, bfrag[nt], acc[1][nt]);
    }
  }

  const int hv = lane >> 4;
  const int nn = lane & 15;
#pragma unroll
  for (int mt = 0; mt < 2; ++mt) {
#pragma unroll
    for (int nt = 0; nt < 4; ++nt) {
      const int nIdx = n0 + nt * 16 + nn;
      const float bv = bias[nIdx];
      const f32x8 av = acc[mt][nt];
#pragma unroll
      for (int r = 0; r < 8; ++r) {
        const int mIdx = m0 + mt * 16 + r + hv * 8;
        float v = av[r] + bv;
        if (MODE == 3) {
          outf[(size_t)mIdx * D_MODEL + nIdx] = v;
        } else {
          const int b = mIdx >> 11, s = mIdx & 2047;
          const int h = nIdx >> 6,  dd = nIdx & 63;
          if (MODE == 0) v *= 0.125f;                 // 1/sqrt(head_dim)
          if (MODE == 2)
            outb[(((size_t)b * NHEADS + h) * HEAD_D + dd) * S_LEN + s] = f2bf(v);
          else
            outb[((((size_t)b * NHEADS + h) * S_LEN + s) << 6) + dd] = f2bf(v);
        }
      }
    }
  }
}

// ---------------------------------------------------------------------------
// Flash attention.  Block = 8 waves on one (b,h); wave owns 16 query rows.
// K/V 64x64 tiles are cooperatively DMA'd into double-buffered LDS via
// global_load_async_to_lds_b128; hand-off = s_wait_asynccnt 0 + barrier.
// Per 64-key chunk: 8 WMMAs for S=Q*K^T, masked online softmax (permlane16
// max butterfly, rowsum via WMMA vs all-ones), P through a private LDS patch
// into A-layout, 8 WMMAs for ctx += P*V.
// ---------------------------------------------------------------------------
__global__ __launch_bounds__(256) void flash_attn_kernel(
    const bf16_t* __restrict__ Qb,   // [B*H][S][64] (pre-scaled)
    const bf16_t* __restrict__ Kb,   // [B*H][S][64]
    const bf16_t* __restrict__ Vt,   // [B*H][64][S]
    const int*    __restrict__ mask, // [B][S]
    bf16_t* __restrict__ Cb) {       // [B][S][1024] bf16 ctx
  __shared__ bf16_t ktile[2][64 * 64];   // 2 x 8 KB
  __shared__ bf16_t vtile[2][64 * 64];   // 2 x 8 KB
  __shared__ bf16_t lds_p[8][16 * 64];   // per-wave P patch, 16 KB

  const int tid  = threadIdx.x;
  const int lane = tid & 31;
  const int wave = tid >> 5;
  const int bh = blockIdx.y;
  const int b = bh >> 4, h = bh & 15;
  const int q0 = blockIdx.x * 128 + wave * 16;

  const bf16_t* Qh = Qb + (size_t)bh * S_LEN * HEAD_D;
  const bf16_t* Kh = Kb + (size_t)bh * S_LEN * HEAD_D;
  const bf16_t* Vh = Vt + (size_t)bh * HEAD_D * S_LEN;
  const int* mrow = mask + b * S_LEN;

  const bf16x16 qf0 = load_frag(Qh, HEAD_D, q0, 0,  lane);
  const bf16x16 qf1 = load_frag(Qh, HEAD_D, q0, 32, lane);

  union { bf16x16 f; unsigned u[8]; } ones;   // B=1s: D = rowsum(P)
#pragma unroll
  for (int i = 0; i < 8; ++i) ones.u[i] = 0x3F803F80u;

  f32x8 ctx[4];
#pragma unroll
  for (int t = 0; t < 4; ++t) ctx[t] = (f32x8){0.f,0.f,0.f,0.f,0.f,0.f,0.f,0.f};
  float m_run[8], l_run[8];
#pragma unroll
  for (int r = 0; r < 8; ++r) { m_run[r] = -1e30f; l_run[r] = 0.f; }

  const int nn = lane & 15;
  const int hv = lane >> 4;
  bf16_t* patch = lds_p[wave];

  // stage one 64x64 tile pair: each thread DMAs 2x16B for K and for V
  auto stage = [&](int k0, int buf) {
#pragma unroll
    for (int j = 0; j < 2; ++j) {
      const int i = tid + j * 256;          // 0..511
      const int row = i >> 3;               // 0..63
      const int c8  = (i & 7) << 3;         // 0,8,..,56
      async_b128((unsigned)(size_t)&ktile[buf][i * 8],
                 Kh + (size_t)(k0 + row) * HEAD_D + c8);
      async_b128((unsigned)(size_t)&vtile[buf][i * 8],
                 Vh + (size_t)row * S_LEN + k0 + c8);
    }
  };

  stage(0, 0);

  const int NIT = S_LEN / 64;
  for (int it = 0; it < NIT; ++it) {
    const int k0 = it * 64;
    const int cur = it & 1;
    asm volatile("s_wait_asynccnt 0" ::: "memory");  // my stage done
    __syncthreads();                                 // everyone's stage done,
                                                     // prev buffer fully read
    if (it + 1 < NIT) stage(k0 + 64, 1 - cur);

    const bf16_t* kt = &ktile[cur][0];
    const bf16_t* vt = &vtile[cur][0];

    // ---- scores ----------------------------------------------------------
    f32x8 sc[4];
#pragma unroll
    for (int t = 0; t < 4; ++t) {
      sc[t] = (f32x8){0.f,0.f,0.f,0.f,0.f,0.f,0.f,0.f};
      sc[t] = wmma_bf16(qf0, load_frag(kt, 64, t * 16, 0,  lane), sc[t]);
      sc[t] = wmma_bf16(qf1, load_frag(kt, 64, t * 16, 32, lane), sc[t]);
    }
    // ---- mask + in-lane row max ------------------------------------------
    bool keep[4];
#pragma unroll
    for (int t = 0; t < 4; ++t) keep[t] = mrow[k0 + t * 16 + nn] != 0;
    float rm[8];
#pragma unroll
    for (int r = 0; r < 8; ++r) {
#pragma unroll
      for (int t = 0; t < 4; ++t) sc[t][r] = keep[t] ? sc[t][r] : NEG_INF;
      rm[r] = fmaxf(fmaxf(sc[0][r], sc[1][r]), fmaxf(sc[2][r], sc[3][r]));
    }
#pragma unroll
    for (int r = 0; r < 8; ++r) rm[r] = halfwave_max16(rm[r]);

    // ---- online softmax --------------------------------------------------
    float scl[8];
#pragma unroll
    for (int r = 0; r < 8; ++r) {
      const float nm = fmaxf(m_run[r], rm[r]);
      scl[r] = __expf(m_run[r] - nm);
      m_run[r] = nm;
#pragma unroll
      for (int t = 0; t < 4; ++t) sc[t][r] = __expf(sc[t][r] - nm);
    }
#pragma unroll
    for (int r = 0; r < 8; ++r) {
      ctx[0][r] *= scl[r]; ctx[1][r] *= scl[r];
      ctx[2][r] *= scl[r]; ctx[3][r] *= scl[r];
    }

    // ---- P: C-layout -> A-layout through private LDS patch ---------------
#pragma unroll
    for (int r = 0; r < 8; ++r) {
      const int row = (r + 8 * hv) * 64 + nn;
      const bf16x2 p01 = __builtin_convertvector((f32x2){sc[0][r], sc[1][r]}, bf16x2);
      const bf16x2 p23 = __builtin_convertvector((f32x2){sc[2][r], sc[3][r]}, bf16x2);
      patch[row]      = p01.x;
      patch[row + 16] = p01.y;
      patch[row + 32] = p23.x;
      patch[row + 48] = p23.y;
    }
    asm volatile("s_wait_dscnt 0" ::: "memory");
    const bf16x16 pf0 = load_frag(patch, 64, 0, 0,  lane);
    const bf16x16 pf1 = load_frag(patch, 64, 0, 32, lane);

    // ---- row-sum of P via WMMA vs all-ones -------------------------------
    f32x8 rs = wmma_bf16(pf0, ones.f, (f32x8){0.f,0.f,0.f,0.f,0.f,0.f,0.f,0.f});
    rs = wmma_bf16(pf1, ones.f, rs);
#pragma unroll
    for (int r = 0; r < 8; ++r) l_run[r] = l_run[r] * scl[r] + rs[r];

    // ---- ctx += P * V ----------------------------------------------------
#pragma unroll
    for (int t = 0; t < 4; ++t) {
      ctx[t] = wmma_bf16(pf0, load_frag(vt, 64, t * 16, 0,  lane), ctx[t]);
      ctx[t] = wmma_bf16(pf1, load_frag(vt, 64, t * 16, 32, lane), ctx[t]);
    }
  }

  float inv[8];
#pragma unroll
  for (int r = 0; r < 8; ++r) inv[r] = 1.f / l_run[r];
#pragma unroll
  for (int t = 0; t < 4; ++t) {
#pragma unroll
    for (int r = 0; r < 8; ++r) {
      const int qIdx = q0 + r + 8 * hv;
      const int dd = t * 16 + nn;
      Cb[((size_t)b * S_LEN + qIdx) * D_MODEL + h * HEAD_D + dd] =
          f2bf(ctx[t][r] * inv[r]);
    }
  }
}

// ---------------------------------------------------------------------------
extern "C" void kernel_launch(void* const* d_in, const int* in_sizes, int n_in,
                              void* d_out, int out_size, void* d_ws, size_t ws_size,
                              hipStream_t stream) {
  (void)in_sizes; (void)n_in; (void)out_size; (void)ws_size;
  const float* query = (const float*)d_in[0];
  const int*   mask  = (const int*)d_in[1];
  const float* Wq = (const float*)d_in[2];
  const float* bq = (const float*)d_in[3];
  const float* Wk = (const float*)d_in[4];
  const float* bk = (const float*)d_in[5];
  const float* Wv = (const float*)d_in[6];
  const float* bv = (const float*)d_in[7];
  const float* Wo = (const float*)d_in[8];
  const float* bo = (const float*)d_in[9];
  float* out = (float*)d_out;

  char* ws = (char*)d_ws;
  const size_t MB = 1ull << 20;
  bf16_t* Xb  = (bf16_t*)(ws);             // 16 MB: query bf16 [8192][1024]
  bf16_t* Wqb = (bf16_t*)(ws + 16 * MB);   // 2 MB each
  bf16_t* Wkb = (bf16_t*)(ws + 18 * MB);
  bf16_t* Wvb = (bf16_t*)(ws + 20 * MB);
  bf16_t* Wob = (bf16_t*)(ws + 22 * MB);
  bf16_t* Qbf = (bf16_t*)(ws + 24 * MB);   // 16 MB [B*H][S][64]
  bf16_t* Kbf = (bf16_t*)(ws + 40 * MB);   // 16 MB [B*H][S][64]
  bf16_t* Vtb = (bf16_t*)(ws + 56 * MB);   // 16 MB [B*H][64][S]
  bf16_t* Cb  = (bf16_t*)(ws + 72 * MB);   // 16 MB ctx bf16 [8192][1024]

  const int nX4 = (8192 * 1024) / 4;
  const int nW4 = (1024 * 1024) / 4;
  cvt_kernel<<<(nX4 + 255) / 256, 256, 0, stream>>>(query, Xb, nX4);
  cvt_kernel<<<(nW4 + 255) / 256, 256, 0, stream>>>(Wq, Wqb, nW4);
  cvt_kernel<<<(nW4 + 255) / 256, 256, 0, stream>>>(Wk, Wkb, nW4);
  cvt_kernel<<<(nW4 + 255) / 256, 256, 0, stream>>>(Wv, Wvb, nW4);
  cvt_kernel<<<(nW4 + 255) / 256, 256, 0, stream>>>(Wo, Wob, nW4);

  dim3 g(64, 8);  // M/128, N/128
  gemm_bf16_kernel<0><<<g, 256, 0, stream>>>(Xb, Wqb, bq, Qbf, nullptr);
  gemm_bf16_kernel<1><<<g, 256, 0, stream>>>(Xb, Wkb, bk, Kbf, nullptr);
  gemm_bf16_kernel<2><<<g, 256, 0, stream>>>(Xb, Wvb, bv, Vtb, nullptr);

  flash_attn_kernel<<<dim3(16, 64), 256, 0, stream>>>(Qbf, Kbf, Vtb, mask, Cb);

  gemm_bf16_kernel<3><<<g, 256, 0, stream>>>(Cb, Wob, bo, nullptr, out);
}
